// GraphPlayerEncoder_55722905699391
// MI455X (gfx1250) — matrix operand order repaired and checked by reading, not compiled, not run on hardware
//
#include <hip/hip_runtime.h>
#include <hip/hip_bf16.h>

// ---------------- problem constants (from reference) ----------------
static constexpr int NN = 22528;          // nodes
static constexpr int GG = 1024;           // graphs
static constexpr int EE = 473088;         // edges
static constexpr int HH = 4;              // heads
static constexpr int LL = 4;              // layers
static constexpr int EP = EE + NN;        // edges + self loops = 495616 (16*30976)

typedef __attribute__((ext_vector_type(16))) _Float16 v16h;
typedef __attribute__((ext_vector_type(8)))  _Float16 v8h;
typedef __attribute__((ext_vector_type(8)))  float    v8f;

// ---------------- WMMA helpers (gfx1250, wave32) ----------------
// D(16x16,f32) += A(16x32,f16) * B(32x16,f16)
__device__ __forceinline__ v8f wmma_f16(v16h a, v16h b, v8f c) {
  return __builtin_amdgcn_wmma_f32_16x16x32_f16(false, a, false, b, (short)0, c,
                                                false, false);
}

// A fragment: row-major A, ld elements per row.
// ISA layout: lanes 0-15 row M=lane, halves 0..7 = K 0..7, halves 8..15 = K 16..23
//             lanes 16-31 row M=lane-16, halves 0..7 = K 8..15, 8..15 = K 24..31
__device__ __forceinline__ v16h load_A(const _Float16* A, int ld, int lane) {
  int r  = lane & 15;
  int kb = (lane >> 4) * 8;
  v8h lo = *(const v8h*)(A + (size_t)r * ld + kb);
  v8h hi = *(const v8h*)(A + (size_t)r * ld + 16 + kb);
  v16h a;
#pragma unroll
  for (int i = 0; i < 8; ++i) { a[i] = lo[i]; a[8 + i] = hi[i]; }
  return a;
}

// B fragment from pre-swizzled tile: tile = 32 lanes x 16 halfs, lane-major.
// Each lane: two contiguous 16B vector loads.
__device__ __forceinline__ v16h load_B_packed(const _Float16* T, int lane) {
  const v8h* p = (const v8h*)(T + lane * 16);
  v8h lo = p[0];
  v8h hi = p[1];
  v16h b;
#pragma unroll
  for (int i = 0; i < 8; ++i) { b[i] = lo[i]; b[8 + i] = hi[i]; }
  return b;
}

// ordered-uint encoding for float atomic max
__device__ __forceinline__ unsigned fenc(float f) {
  unsigned u = __float_as_uint(f);
  return (u & 0x80000000u) ? ~u : (u | 0x80000000u);
}
__device__ __forceinline__ float fdec(unsigned u) {
  return __uint_as_float((u & 0x80000000u) ? (u & 0x7FFFFFFFu) : ~u);
}

// ---------------- small utility kernels ----------------
__global__ void k_fill_f32(float* p, float v, int n) {
  int t = blockIdx.x * blockDim.x + threadIdx.x;
  if (t < n) p[t] = v;
}
__global__ void k_fill_u32(unsigned* p, unsigned v, int n) {
  int t = blockIdx.x * blockDim.x + threadIdx.x;
  if (t < n) p[t] = v;
}

// ---- weight conversion into WMMA fragment order ----
// Tile layout for a (32 x 16) K-slab of B: 32 lanes x 16 halfs contiguous per lane.
// Lane L covers col n = L&15, K rows koff+(i&7)+(i>>3)*16 where koff = (L>>4)*8.
// Proj weights: combined [Wl|Wr] viewed as (64 K x 512 cols) per layer ->
//   tiles indexed (kc in 0..1, ct in 0..31), tile offset (kc*32+ct)*512.
__global__ void k_cvt_proj(const float* Wl, const float* Wr, _Float16* Wp) {
  int t = blockIdx.x * blockDim.x + threadIdx.x;
  if (t >= LL * 2 * 32 * 32 * 16) return;
  int i    = t & 15;
  int lane = (t >> 4) & 31;
  int ct   = (t >> 9) & 31;
  int kc   = (t >> 14) & 1;
  int l    = t >> 15;
  int n    = lane & 15;
  int koff = (lane >> 4) * 8;
  int K    = kc * 32 + koff + (i & 7) + (i >> 3) * 16;        // 0..63
  int col  = ct * 16 + n;               // 0..511
  float v = (col < 256) ? Wl[((size_t)l * 64 + K) * 256 + col]
                        : Wr[((size_t)l * 64 + K) * 256 + (col - 256)];
  Wp[t] = (_Float16)v;
}

// Social weights: W1 (128x64): 4 kc x 4 ct tiles at offset 0 (8192 halfs);
// W2 (64x64): 2 kc x 4 ct at 8192; Wg (64x64): 2 kc x 4 ct at 12288.
__global__ void k_cvt_sp(const float* W1, const float* W2, const float* Wg, _Float16* o) {
  int t = blockIdx.x * blockDim.x + threadIdx.x;
  if (t >= 16384) return;
  int u, kc;
  const float* W;
  if (t < 8192)       { u = t;         kc = u >> 11; W = W1; }
  else if (t < 12288) { u = t - 8192;  kc = u >> 11; W = W2; }
  else                { u = t - 12288; kc = u >> 11; W = Wg; }
  int i    = u & 15;
  int lane = (u >> 4) & 31;
  int ct   = (u >> 9) & 3;
  int n    = lane & 15;
  int koff = (lane >> 4) * 8;
  int K    = kc * 32 + koff + (i & 7) + (i >> 3) * 16;
  int col  = ct * 16 + n;
  o[t] = (_Float16)W[(size_t)K * 64 + col];
}

// ---------------- embedding ----------------
__global__ void k_ctx(const float* ctx, const int* formation, const int* alig,
                      const float* W_ctx, const float* b_ctx,
                      const float* form_t, const float* align_t, float* cbuf) {
  int t = blockIdx.x * blockDim.x + threadIdx.x;
  if (t >= GG * 64) return;
  int g = t >> 6, k = t & 63;
  float acc = b_ctx[k];
#pragma unroll
  for (int j = 0; j < 3; ++j) acc += ctx[g * 3 + j] * W_ctx[j * 64 + k];
  acc = acc > 0.f ? acc : 0.f;
  acc += form_t[formation[g] * 64 + k] + align_t[alig[g] * 64 + k];
  cbuf[t] = acc;
}

__global__ void k_embed(const float* x, const int* role, const int* side,
                        const int* batch, const float* W_emb, const float* b_emb,
                        const float* role_t, const float* side_t,
                        const float* cbuf, float* h, _Float16* hf) {
  int t = blockIdx.x * blockDim.x + threadIdx.x;
  if (t >= NN * 64) return;
  int n = t >> 6, k = t & 63;
  float acc = b_emb[k];
#pragma unroll
  for (int j = 0; j < 7; ++j) acc += x[n * 7 + j] * W_emb[j * 64 + k];
  acc = acc > 0.f ? acc : 0.f;
  acc += role_t[role[n] * 64 + k];
  if (k < 32) acc += side_t[side[n] * 32 + k];
  acc += cbuf[(size_t)batch[n] * 64 + k];
  h[t] = acc;
  hf[t] = (_Float16)acc;
}

// ---------------- self-loop edge_attr prep ----------------
__global__ void k_deg(const int* dst, const float* eattr, float* degf, float* selfattr) {
  int e = blockIdx.x * blockDim.x + threadIdx.x;
  if (e >= EE) return;
  int d = dst[e];
  atomicAdd(&degf[d], 1.0f);
#pragma unroll
  for (int j = 0; j < 5; ++j) atomicAdd(&selfattr[(size_t)d * 5 + j], eattr[(size_t)e * 5 + j]);
}
__global__ void k_selfdiv(float* degf, float* selfattr) {
  int n = blockIdx.x * blockDim.x + threadIdx.x;
  if (n >= NN) return;
  float d = fmaxf(degf[n], 1.0f);
#pragma unroll
  for (int j = 0; j < 5; ++j) selfattr[(size_t)n * 5 + j] /= d;
}

// ---------------- GAT projection: [XL|XR] = h @ [Wl|Wr] + [bl|br] ----------------
// grid (1408, 4), block 256 (8 waves); wave computes one 16x16 tile of the N x 512 output.
__global__ void k_proj_wmma(const _Float16* hf16, const _Float16* Wp,
                            const float* bl, const float* br, float* XL, float* XR) {
  int lane = threadIdx.x & 31, wave = threadIdx.x >> 5;
  int mtile = blockIdx.x;
  int ctile = blockIdx.y * 8 + wave;  // 0..31 over 512 cols
  const _Float16* A = hf16 + (size_t)mtile * 16 * 64;
  v8f acc = {};
#pragma unroll
  for (int kc = 0; kc < 2; ++kc)
    acc = wmma_f16(load_A(A + kc * 32, 64, lane),
                   load_B_packed(Wp + (size_t)(kc * 32 + ctile) * 512, lane), acc);
  int col = ctile * 16 + (lane & 15);
  int rbase = mtile * 16 + (lane >> 4) * 8;
  float bias;
  float* D;
  int c2;
  if (col < 256) { bias = bl[col]; D = XL; c2 = col; }
  else           { bias = br[col - 256]; D = XR; c2 = col - 256; }
#pragma unroll
  for (int v = 0; v < 8; ++v) D[(size_t)(rbase + v) * 256 + c2] = acc[v] + bias;
}

// ---------------- attention: logits + segment max ----------------
__global__ void k_edge_logits(const float* XL, const float* XR, const int* src,
                              const int* dst, const float* eattr, const float* selfattr,
                              const float* We, const float* att, float* logits,
                              unsigned* menc) {
  int t = blockIdx.x * blockDim.x + threadIdx.x;
  if (t >= EP * HH) return;
  int e = t >> 2, hh = t & 3;
  int s, d;
  const float* ap;
  if (e < EE) { s = src[e]; d = dst[e]; ap = eattr + (size_t)e * 5; }
  else        { s = d = e - EE;         ap = selfattr + (size_t)(e - EE) * 5; }
  float a0 = ap[0], a1 = ap[1], a2 = ap[2], a3 = ap[3], a4 = ap[4];
  const float* xl = XL + (size_t)s * 256 + hh * 64;
  const float* xr = XR + (size_t)d * 256 + hh * 64;
  const float* we = We + hh * 64;  // rows stride 256
  const float* at = att + hh * 64;
  float acc = 0.f;
#pragma unroll 4
  for (int c = 0; c < 64; ++c) {
    float ee = a0 * we[c] + a1 * we[256 + c] + a2 * we[512 + c] +
               a3 * we[768 + c] + a4 * we[1024 + c];
    float v = xl[c] + xr[c] + ee;
    v = v > 0.f ? v : 0.2f * v;  // leaky_relu 0.2
    acc += v * at[c];
  }
  logits[t] = acc;
  atomicMax(&menc[(size_t)d * HH + hh], fenc(acc));
}

// p = exp(logit - m[dst]); segment-sum denom
__global__ void k_edge_p(const int* dst, float* logits, const unsigned* menc,
                         float* denom) {
  int t = blockIdx.x * blockDim.x + threadIdx.x;
  if (t >= EP * HH) return;
  int e = t >> 2, hh = t & 3;
  int d = (e < EE) ? dst[e] : (e - EE);
  float pv = __expf(logits[t] - fdec(menc[(size_t)d * HH + hh]));
  logits[t] = pv;
  atomicAdd(&denom[(size_t)d * HH + hh], pv);
}

// out[dst] += sum_h alpha_h * x_l[src,h,:] / H   (warp per edge, 2 cols per lane)
__global__ void k_edge_accum(const float* XL, const int* src, const int* dst,
                             const float* p, const float* denom, float* outacc) {
  int warp = (blockIdx.x * blockDim.x + threadIdx.x) >> 5;
  int lane = threadIdx.x & 31;
  if (warp >= EP) return;
  int e = warp, s, d;
  if (e < EE) { s = src[e]; d = dst[e]; } else { s = d = e - EE; }
  float al[4];
#pragma unroll
  for (int hh = 0; hh < 4; ++hh)
    al[hh] = p[(size_t)e * 4 + hh] / denom[(size_t)d * 4 + hh] * 0.25f;  // /H folded in
  const float* xl = XL + (size_t)s * 256;
  float s0 = 0.f, s1 = 0.f;
#pragma unroll
  for (int hh = 0; hh < 4; ++hh) {
    s0 += al[hh] * xl[hh * 64 + lane];
    s1 += al[hh] * xl[hh * 64 + lane + 32];
  }
  atomicAdd(&outacc[(size_t)d * 64 + lane], s0);
  atomicAdd(&outacc[(size_t)d * 64 + lane + 32], s1);
}

// h = LN(relu(out + gat_bias) + h_res)   (warp per node); also refresh f16 copy
__global__ void k_node_update(float* h, _Float16* hf, const float* outacc,
                              const float* gbias, const float* g, const float* b) {
  int node = blockIdx.x * (blockDim.x >> 5) + (threadIdx.x >> 5);
  int lane = threadIdx.x & 31;
  if (node >= NN) return;
  float r0 = outacc[(size_t)node * 64 + lane] + gbias[lane];
  float r1 = outacc[(size_t)node * 64 + lane + 32] + gbias[lane + 32];
  r0 = r0 > 0.f ? r0 : 0.f;
  r1 = r1 > 0.f ? r1 : 0.f;
  float v0 = r0 + h[(size_t)node * 64 + lane];
  float v1 = r1 + h[(size_t)node * 64 + lane + 32];
  float s = v0 + v1;
#pragma unroll
  for (int o = 16; o > 0; o >>= 1) s += __shfl_down(s, o, 32);
  s = __shfl(s, 0, 32);
  float mean = s * (1.0f / 64.0f);
  float d0 = v0 - mean, d1 = v1 - mean;
  float q = d0 * d0 + d1 * d1;
#pragma unroll
  for (int o = 16; o > 0; o >>= 1) q += __shfl_down(q, o, 32);
  q = __shfl(q, 0, 32);
  float rstd = rsqrtf(q * (1.0f / 64.0f) + 1e-5f);
  float o0 = d0 * rstd * g[lane] + b[lane];
  float o1 = d1 * rstd * g[lane + 32] + b[lane + 32];
  h[(size_t)node * 64 + lane]       = o0;
  h[(size_t)node * 64 + lane + 32]  = o1;
  hf[(size_t)node * 64 + lane]      = (_Float16)o0;
  hf[(size_t)node * 64 + lane + 32] = (_Float16)o1;
}

// ---------------- social pooling: 3 chained WMMAs per 16-edge tile ----------------
// block = 128 threads (4 waves); wave owns one 16-col tile of the 64-wide output.
// Pair gather uses GLOBAL_LOAD_ASYNC_TO_LDS_B128 (ASYNCcnt-tracked, no VGPR bounce).
__global__ void k_social_wmma(const _Float16* hf16, const int* src, const int* dst,
                              const _Float16* W1p, const float* b1,
                              const _Float16* W2p, const float* b2,
                              const _Float16* Wgp, const float* bg,
                              float* pooled, float* cntb) {
  __shared__ _Float16 pairA[16 * 128];
  __shared__ _Float16 act[16 * 64];
  __shared__ _Float16 t16[16 * 64];
  int tid = threadIdx.x, lane = tid & 31, wave = tid >> 5;
  int e0 = blockIdx.x * 16;
  // async gather pair = [h[src] | h[dst]] into LDS, 128-bit per lane
  for (int i = tid; i < 16 * 16; i += 128) {
    int row = i >> 4, ch = i & 15;   // 16 chunks of 8 halfs per row
    int e = e0 + row;
    int node = (ch < 8) ? src[e] : dst[e];
    const _Float16* gsrc = hf16 + (size_t)node * 64 + (ch & 7) * 8;
    unsigned ldsa = (unsigned)(size_t)(pairA + row * 128 + ch * 8);
    asm volatile("global_load_async_to_lds_b128 %0, %1, off"
                 :
                 : "v"(ldsa), "v"(gsrc)
                 : "memory");
  }
  if (tid < 16) atomicAdd(&cntb[src[e0 + tid]], 1.0f);
  asm volatile("s_wait_asynccnt 0" ::: "memory");
  __syncthreads();

  int col = wave * 16 + (lane & 15);
  int rbase = (lane >> 4) * 8;

  // stage 1: relu(pair @ W1 + b1)
  v8f acc1 = {};
#pragma unroll
  for (int kc = 0; kc < 4; ++kc)
    acc1 = wmma_f16(load_A(pairA + kc * 32, 128, lane),
                    load_B_packed(W1p + (size_t)(kc * 4 + wave) * 512, lane), acc1);
  float bb1 = b1[col];
#pragma unroll
  for (int v = 0; v < 8; ++v) {
    float xx = acc1[v] + bb1;
    xx = xx > 0.f ? xx : 0.f;
    act[(rbase + v) * 64 + col] = (_Float16)xx;
  }
  __syncthreads();

  // stage 2: inter = act @ W2 + b2
  v8f acc2 = {};
#pragma unroll
  for (int kc = 0; kc < 2; ++kc)
    acc2 = wmma_f16(load_A(act + kc * 32, 64, lane),
                    load_B_packed(W2p + (size_t)(kc * 4 + wave) * 512, lane), acc2);
  float tval[8];
  float bb2 = b2[col];
#pragma unroll
  for (int v = 0; v < 8; ++v) {
    tval[v] = acc2[v] + bb2;
    t16[(rbase + v) * 64 + col] = (_Float16)tval[v];
  }
  __syncthreads();

  // stage 3: gated = inter * sigmoid(inter @ Wg + bg), scatter onto src
  v8f acc3 = {};
#pragma unroll
  for (int kc = 0; kc < 2; ++kc)
    acc3 = wmma_f16(load_A(t16 + kc * 32, 64, lane),
                    load_B_packed(Wgp + (size_t)(kc * 4 + wave) * 512, lane), acc3);
  float bbg = bg[col];
#pragma unroll
  for (int v = 0; v < 8; ++v) {
    float gate = 1.0f / (1.0f + __expf(-(acc3[v] + bbg)));
    float out = tval[v] * gate;
    int e = e0 + rbase + v;
    atomicAdd(&pooled[(size_t)src[e] * 64 + col], out);
  }
}

// final: out = LN(h + pooled/max(cnt,1))
__global__ void k_final_ln(const float* h, const float* pooled, const float* cntb,
                           const float* g, const float* b, float* out) {
  int node = blockIdx.x * (blockDim.x >> 5) + (threadIdx.x >> 5);
  int lane = threadIdx.x & 31;
  if (node >= NN) return;
  float c = fmaxf(cntb[node], 1.0f);
  float v0 = h[(size_t)node * 64 + lane]      + pooled[(size_t)node * 64 + lane] / c;
  float v1 = h[(size_t)node * 64 + lane + 32] + pooled[(size_t)node * 64 + lane + 32] / c;
  float s = v0 + v1;
#pragma unroll
  for (int o = 16; o > 0; o >>= 1) s += __shfl_down(s, o, 32);
  s = __shfl(s, 0, 32);
  float mean = s * (1.0f / 64.0f);
  float d0 = v0 - mean, d1 = v1 - mean;
  float q = d0 * d0 + d1 * d1;
#pragma unroll
  for (int o = 16; o > 0; o >>= 1) q += __shfl_down(q, o, 32);
  q = __shfl(q, 0, 32);
  float rstd = rsqrtf(q * (1.0f / 64.0f) + 1e-5f);
  out[(size_t)node * 64 + lane]      = d0 * rstd * g[lane] + b[lane];
  out[(size_t)node * 64 + lane + 32] = d1 * rstd * g[lane + 32] + b[lane + 32];
}

// ---------------- launch ----------------
extern "C" void kernel_launch(void* const* d_in, const int* in_sizes, int n_in,
                              void* d_out, int out_size, void* d_ws, size_t ws_size,
                              hipStream_t stream) {
  const float* x         = (const float*)d_in[0];
  const int*   eidx      = (const int*)d_in[1];
  const float* eattr     = (const float*)d_in[2];
  const float* context   = (const float*)d_in[3];
  const int*   batch     = (const int*)d_in[4];
  const int*   role      = (const int*)d_in[5];
  const int*   side      = (const int*)d_in[6];
  const int*   formation = (const int*)d_in[7];
  const int*   alig      = (const int*)d_in[8];
  const float* W_emb     = (const float*)d_in[9];
  const float* b_emb     = (const float*)d_in[10];
  const float* role_t    = (const float*)d_in[11];
  const float* side_t    = (const float*)d_in[12];
  const float* W_ctx     = (const float*)d_in[13];
  const float* b_ctx     = (const float*)d_in[14];
  const float* form_t    = (const float*)d_in[15];
  const float* align_t   = (const float*)d_in[16];
  const float* Wl        = (const float*)d_in[17];
  const float* bl        = (const float*)d_in[18];
  const float* Wr        = (const float*)d_in[19];
  const float* br        = (const float*)d_in[20];
  const float* We        = (const float*)d_in[21];
  const float* att       = (const float*)d_in[22];
  const float* gat_bias  = (const float*)d_in[23];
  const float* ln_g      = (const float*)d_in[24];
  const float* ln_b      = (const float*)d_in[25];
  const float* spW1      = (const float*)d_in[26];
  const float* spb1      = (const float*)d_in[27];
  const float* spW2      = (const float*)d_in[28];
  const float* spb2      = (const float*)d_in[29];
  const float* spWg      = (const float*)d_in[30];
  const float* spbg      = (const float*)d_in[31];
  const float* fn_g      = (const float*)d_in[32];
  const float* fn_b      = (const float*)d_in[33];
  const int* src = eidx;
  const int* dstp = eidx + EE;

  // workspace carve-out (256B aligned)
  size_t off = 0;
  char* base = (char*)d_ws;
  auto carve = [&](size_t bytes) -> char* {
    char* p = base + off;
    off += (bytes + 255) & ~(size_t)255;
    return p;
  };
  float*     hbuf    = (float*)carve((size_t)NN * 64 * 4);
  _Float16*  hf16    = (_Float16*)carve((size_t)NN * 64 * 2);
  float*     cbuf    = (float*)carve((size_t)GG * 64 * 4);
  float*     XL      = (float*)carve((size_t)NN * 256 * 4);
  float*     XR      = (float*)carve((size_t)NN * 256 * 4);
  _Float16*  Wp16    = (_Float16*)carve((size_t)LL * 64 * 512 * 2);
  _Float16*  spw16   = (_Float16*)carve((size_t)16384 * 2);
  float*     selfat  = (float*)carve((size_t)NN * 5 * 4);
  float*     degf    = (float*)carve((size_t)NN * 4);
  float*     logitsP = (float*)carve((size_t)EP * HH * 4);
  unsigned*  menc    = (unsigned*)carve((size_t)NN * HH * 4);
  float*     denom   = (float*)carve((size_t)NN * HH * 4);
  float*     outacc  = (float*)carve((size_t)NN * 64 * 4);
  float*     pooled  = (float*)carve((size_t)NN * 64 * 4);
  float*     cntb    = (float*)carve((size_t)NN * 4);
  _Float16*  W1p = spw16;
  _Float16*  W2p = spw16 + 8192;
  _Float16*  Wgp = spw16 + 12288;

  const int TB = 256;
  auto blks = [](long n, int tb) { return (int)((n + tb - 1) / tb); };

  // self-loop attr (mean of incoming edge_attr)
  k_fill_f32<<<blks(NN, TB), TB, 0, stream>>>(degf, 0.f, NN);
  k_fill_f32<<<blks((long)NN * 5, TB), TB, 0, stream>>>(selfat, 0.f, NN * 5);
  k_deg<<<blks(EE, TB), TB, 0, stream>>>(dstp, eattr, degf, selfat);
  k_selfdiv<<<blks(NN, TB), TB, 0, stream>>>(degf, selfat);

  // embeddings (writes h in f32 and f16)
  k_ctx<<<blks((long)GG * 64, TB), TB, 0, stream>>>(context, formation, alig, W_ctx,
                                                    b_ctx, form_t, align_t, cbuf);
  k_embed<<<blks((long)NN * 64, TB), TB, 0, stream>>>(x, role, side, batch, W_emb,
                                                      b_emb, role_t, side_t, cbuf,
                                                      hbuf, hf16);

  // weight conversion to f16 fragment order
  k_cvt_proj<<<blks((long)LL * 64 * 512, TB), TB, 0, stream>>>(Wl, Wr, Wp16);
  k_cvt_sp<<<blks(16384, TB), TB, 0, stream>>>(spW1, spW2, spWg, spw16);

  // GAT layers
  for (int l = 0; l < LL; ++l) {
    k_proj_wmma<<<dim3(NN / 16, 4), TB, 0, stream>>>(
        hf16, Wp16 + (size_t)l * 64 * 512, bl + (size_t)l * 256, br + (size_t)l * 256,
        XL, XR);
    k_fill_u32<<<blks((long)NN * HH, TB), TB, 0, stream>>>(menc, 0u, NN * HH);
    k_fill_f32<<<blks((long)NN * HH, TB), TB, 0, stream>>>(denom, 0.f, NN * HH);
    k_fill_f32<<<blks((long)NN * 64, TB), TB, 0, stream>>>(outacc, 0.f, NN * 64);
    k_edge_logits<<<blks((long)EP * HH, TB), TB, 0, stream>>>(
        XL, XR, src, dstp, eattr, selfat, We + (size_t)l * 5 * 256,
        att + (size_t)l * 256, logitsP, menc);
    k_edge_p<<<blks((long)EP * HH, TB), TB, 0, stream>>>(dstp, logitsP, menc, denom);
    k_edge_accum<<<EP / 8, TB, 0, stream>>>(XL, src, dstp, logitsP, denom, outacc);
    k_node_update<<<NN / 8, TB, 0, stream>>>(hbuf, hf16, outacc,
                                             gat_bias + (size_t)l * 64,
                                             ln_g + (size_t)l * 64,
                                             ln_b + (size_t)l * 64);
  }

  // social pooling
  k_fill_f32<<<blks((long)NN * 64, TB), TB, 0, stream>>>(pooled, 0.f, NN * 64);
  k_fill_f32<<<blks(NN, TB), TB, 0, stream>>>(cntb, 0.f, NN);
  k_social_wmma<<<EE / 16, 128, 0, stream>>>(hf16, src, dstp, W1p, spb1, W2p, spb2,
                                             Wgp, spbg, pooled, cntb);
  k_final_ln<<<NN / 8, TB, 0, stream>>>(hbuf, pooled, cntb, fn_g, fn_b, (float*)d_out);
}
